// Discriminator_5549097747099
// MI455X (gfx1250) — matrix operand order repaired and checked by reading, not compiled, not run on hardware
//
#include <hip/hip_runtime.h>
#include <hip/hip_bf16.h>
#include <math.h>

typedef _Float16 half_t;
typedef __attribute__((ext_vector_type(2)))  _Float16 h2;
typedef __attribute__((ext_vector_type(16))) _Float16 v16h;
typedef __attribute__((ext_vector_type(8)))  float    v8f;

__device__ __constant__ float kF2[3][3] = {
  {1.f/16.f, 2.f/16.f, 1.f/16.f},
  {2.f/16.f, 4.f/16.f, 2.f/16.f},
  {1.f/16.f, 2.f/16.f, 1.f/16.f}};

__device__ __forceinline__ float lrelu_f(float v){
  return (v < 0.f ? 0.2f*v : v) * 1.3867504905630728f; // sqrt(2/(1+0.2^2))
}

// ---------------------------------------------------------------------------
// WMMA GEMM: out[M,Nc] = A[M,K] (fp16) x Bw[Nc,K]^T (fp16, weight-native O-major)
// One wave = 32x64 tile: 2 A-frags x 4 B-frags -> 8 independent accumulators.
// blockDim=128 (4 waves over M) => block covers 128 rows x 64 cols.
// K is a template constant so all intra-tile displacements become instruction
// immediates: only two base pointers (A,B) live in VGPRs -> no spills.
// Requires M % 128 == 0, Nc % 64 == 0, K % 32 == 0 (true for all layers here).
// ---------------------------------------------------------------------------
template<int K>
__global__ void k_gemm_wmma(const half_t* __restrict__ A, const half_t* __restrict__ Bw,
                            const float* __restrict__ bias, const float* __restrict__ resid,
                            float* __restrict__ outF, half_t* __restrict__ outH,
                            int M, int Nc, int act)
{
  const int lane = threadIdx.x & 31;
  const int wave = threadIdx.x >> 5;
  const int n0 = blockIdx.x * 64;
  const int m0 = blockIdx.y * 128 + wave * 32;
  const int r16  = lane & 15;
  const int hsel = lane >> 4;
  // A frag rows (ISA 16-bit A 16x32 layout): elements 0..7 -> K=k0+hsel*8+i,
  // elements 8..15 -> K=k0+16+hsel*8+i
  const half_t* arow = A + (size_t)(m0 + r16) * K + hsel * 8;
  // B frag cols: lane holds column n0+ni*16+r16; same K layout as A
  const half_t* brow = Bw + (size_t)(n0 + r16) * K + hsel * 8;

  v8f acc[2][4];
#pragma unroll
  for (int mi = 0; mi < 2; ++mi)
#pragma unroll
    for (int ni = 0; ni < 4; ++ni) acc[mi][ni] = (v8f){};

  for (int k0 = 0; k0 < K; k0 += 32) {
    union FragU { uint4 q[2]; v16h v; };
    FragU fa[2], fb[4];
#pragma unroll
    for (int mi = 0; mi < 2; ++mi) {
      const half_t* a = arow + k0 + mi * (16 * K);   // compile-time displacement
      fa[mi].q[0] = *(const uint4*)(a);
      fa[mi].q[1] = *(const uint4*)(a + 16);
    }
#pragma unroll
    for (int ni = 0; ni < 4; ++ni) {
      const half_t* b = brow + k0 + ni * (16 * K);   // compile-time displacement
      fb[ni].q[0] = *(const uint4*)(b);
      fb[ni].q[1] = *(const uint4*)(b + 16);
    }
#pragma unroll
    for (int mi = 0; mi < 2; ++mi)
#pragma unroll
      for (int ni = 0; ni < 4; ++ni)
        acc[mi][ni] = __builtin_amdgcn_wmma_f32_16x16x32_f16(
            false, fa[mi].v, false, fb[ni].v, (short)0, acc[mi][ni], false, false);
  }

#pragma unroll
  for (int mi = 0; mi < 2; ++mi) {
    const int mBase = m0 + mi * 16 + hsel * 8;
#pragma unroll
    for (int ni = 0; ni < 4; ++ni) {
      const int nOut = n0 + ni * 16 + r16;
      const float bval = bias ? bias[nOut] : 0.f;
#pragma unroll
      for (int j = 0; j < 8; ++j) {
        float v = acc[mi][ni][j] + bval;
        if (act) v = lrelu_f(v);
        size_t idx = (size_t)(mBase + j) * Nc + nOut;
        if (outH) outH[idx] = (half_t)v;
        if (outF) outF[idx] = resid ? (resid[idx] + v) : v;
      }
    }
  }
}

// host-side dispatch over the four K values used by this network
static void launch_gemm(int K, dim3 grid, hipStream_t stream,
                        const half_t* A, const half_t* Bw, const float* bias,
                        const float* resid, float* outF, half_t* outH,
                        int M, int Nc, int act)
{
  switch (K) {
    case 128:  k_gemm_wmma<128><<<grid, 128, 0, stream>>>(A, Bw, bias, resid, outF, outH, M, Nc, act); break;
    case 256:  k_gemm_wmma<256><<<grid, 128, 0, stream>>>(A, Bw, bias, resid, outF, outH, M, Nc, act); break;
    case 512:  k_gemm_wmma<512><<<grid, 128, 0, stream>>>(A, Bw, bias, resid, outF, outH, M, Nc, act); break;
    case 1024: k_gemm_wmma<1024><<<grid, 128, 0, stream>>>(A, Bw, bias, resid, outF, outH, M, Nc, act); break;
    default: break;
  }
}

// ---------------------------------------------------------------------------
// Weight prep (1x1): center over I per output row, optional per-row gain, fp16 out.
// ---------------------------------------------------------------------------
__global__ void k_prep_pw(const float* __restrict__ w, const float* __restrict__ rowGain,
                          half_t* __restrict__ out, int I)
{
  const int o = blockIdx.x;
  const float* wr = w + (size_t)o * I;
  __shared__ float red[256];
  float s = 0.f;
  for (int i = threadIdx.x; i < I; i += 256) s += wr[i];
  red[threadIdx.x] = s; __syncthreads();
  for (int st = 128; st > 0; st >>= 1) {
    if (threadIdx.x < st) red[threadIdx.x] += red[threadIdx.x + st];
    __syncthreads();
  }
  const float mean = red[0] / (float)I;
  const float g = rowGain ? rowGain[o] : 1.f;
  for (int i = threadIdx.x; i < I; i += 256)
    out[(size_t)o * I + i] = (half_t)((wr[i] - mean) * g);
}

// Grouped-conv weight prep: center over cpg*9, permute [o][ci][tap] -> [o][tap][ci]
// so the conv inner loop can use packed f16 loads over ci.
__global__ void k_prep_gconv(const float* __restrict__ w, half_t* __restrict__ out, int cpg)
{
  const int o = blockIdx.x;
  const int I = cpg * 9;
  const float* wr = w + (size_t)o * I;
  __shared__ float red[256];
  float s = 0.f;
  for (int i = threadIdx.x; i < I; i += 256) s += wr[i];
  red[threadIdx.x] = s; __syncthreads();
  for (int st = 128; st > 0; st >>= 1) {
    if (threadIdx.x < st) red[threadIdx.x] += red[threadIdx.x + st];
    __syncthreads();
  }
  const float mean = red[0] / (float)I;
  for (int i = threadIdx.x; i < I; i += 256) {
    int ci = i / 9, tap = i % 9;
    out[(size_t)o * I + tap * cpg + ci] = (half_t)(wr[i] - mean);
  }
}

__global__ void k_prep_extract(const float* __restrict__ w, const float* __restrict__ gain,
                               float* __restrict__ out)
{
  const int o = blockIdx.x;
  __shared__ float mean;
  if (threadIdx.x == 0) {
    float s = 0.f;
    for (int i = 0; i < 27; ++i) s += w[o * 27 + i];
    mean = s / 27.f;
  }
  __syncthreads();
  for (int i = threadIdx.x; i < 27; i += blockDim.x) {
    int ci = i / 9;
    out[o * 27 + i] = (w[o * 27 + i] - mean) * gain[ci];
  }
}

__global__ void k_prep_headw(const float* __restrict__ w, float* __restrict__ out)
{
  __shared__ float red[256];
  float s = 0.f;
  for (int i = threadIdx.x; i < 2048; i += 256) s += w[i];
  red[threadIdx.x] = s; __syncthreads();
  for (int st = 128; st > 0; st >>= 1) {
    if (threadIdx.x < st) red[threadIdx.x] += red[threadIdx.x + st];
    __syncthreads();
  }
  float mean = red[0] / 2048.f;
  for (int i = threadIdx.x; i < 2048; i += 256) out[i] = w[i] - mean;
}

// alpha = tanh(ap); gin1 = rsqrt(1+a0^2); gds = rsqrt(1+a0^2[+a1^2])
__global__ void k_gins(const float* __restrict__ ap0, const float* __restrict__ ap1,
                       float* __restrict__ a0, float* __restrict__ a1,
                       float* __restrict__ g1, float* __restrict__ gds, int C)
{
  int c = blockIdx.x * blockDim.x + threadIdx.x;
  if (c >= C) return;
  float t0 = tanhf(ap0[c]);
  a0[c] = t0;
  float av = 1.f + t0 * t0;
  g1[c] = rsqrtf(av);
  if (ap1) { float t1 = tanhf(ap1[c]); a1[c] = t1; gds[c] = rsqrtf(av + t1 * t1); }
  else     { gds[c] = rsqrtf(av); }
}

// bias-add + centered 3x3 extract conv: NCHW f32 in -> NHWC f32 out (Co=128)
__global__ void k_extract(const float* __restrict__ xin, const float* __restrict__ bias,
                          const float* __restrict__ wc, float* __restrict__ out,
                          int total)
{
  int idx = blockIdx.x * blockDim.x + threadIdx.x;
  if (idx >= total) return;
  const int o = idx & 127;
  const int t = idx >> 7;
  const int p = t & 4095;
  const int n = t >> 12;
  const int y = p >> 6, x = p & 63;
  float acc = 0.f;
  for (int ci = 0; ci < 3; ++ci) {
    for (int dy = 0; dy < 3; ++dy) {
      int yy = y + dy - 1; if (yy < 0 || yy >= 64) continue;
      for (int dx = 0; dx < 3; ++dx) {
        int xx = x + dx - 1; if (xx < 0 || xx >= 64) continue;
        float v = xin[((size_t)(n * 3 + ci) * 64 + yy) * 64 + xx] + bias[ci];
        acc += wc[((o * 3 + ci) * 3 + dy) * 3 + dx] * v;
      }
    }
  }
  out[idx] = acc;
}

// NHWC f32 -> fp16 with optional per-channel gain
__global__ void k_tohalf(const float* __restrict__ X, const float* __restrict__ gin,
                         half_t* __restrict__ A, int total, int C)
{
  int idx = blockIdx.x * blockDim.x + threadIdx.x;
  if (idx >= total) return;
  int c = idx % C;
  float g = gin ? gin[c] : 1.f;
  A[idx] = (half_t)(X[idx] * g);
}

// grouped 3x3 conv + lrelu, NHWC fp16 -> fp16; packed-f16 inner loop over ci
// weights pre-permuted to [o][tap][ci]
__global__ void k_gconv(const half_t* __restrict__ in, const half_t* __restrict__ w,
                        half_t* __restrict__ out, int M, int H, int Wd, int hid, int cpg)
{
  int idx = blockIdx.x * blockDim.x + threadIdx.x;
  if (idx >= M * hid) return;
  const int o = idx % hid;
  const int tok = idx / hid;
  const int g = o / cpg;
  const int L = H * Wd;
  const int p = tok % L;
  const int y = p / Wd, x = p % Wd;
  const int cp2 = cpg >> 1;
  h2 acc2 = (h2)0;
  for (int dy = 0; dy < 3; ++dy) {
    int yy = y + dy - 1; if (yy < 0 || yy >= H) continue;
    for (int dx = 0; dx < 3; ++dx) {
      int xx = x + dx - 1; if (xx < 0 || xx >= Wd) continue;
      const h2* ip = (const h2*)(in + ((size_t)tok + (dy - 1) * Wd + (dx - 1)) * hid + g * cpg);
      const h2* wp = (const h2*)(w + ((size_t)o * cpg) * 9 + (dy * 3 + dx) * cpg);
      for (int q = 0; q < cp2; ++q)
        acc2 = ip[q] * wp[q] + acc2;   // packed v_pk_fma_f16
    }
  }
  float acc = (float)acc2.x + (float)acc2.y;
  out[(size_t)tok * hid + o] = (half_t)lrelu_f(acc);
}

// depthwise blur, stride 2, same channel count, with per-channel pre-gain
__global__ void k_ds_same(const float* __restrict__ in, const float* __restrict__ gin,
                          float* __restrict__ out, int total, int H, int Wd, int C)
{
  int idx = blockIdx.x * blockDim.x + threadIdx.x;
  if (idx >= total) return;
  const int Ho = H / 2, Wo = Wd / 2;
  int c = idx % C;
  int t = idx / C;
  int xo = t % Wo; t /= Wo;
  int yo = t % Ho;
  int n  = t / Ho;
  float acc = 0.f;
  for (int dy = 0; dy < 3; ++dy) {
    int yy = 2 * yo + dy - 1; if (yy < 0 || yy >= H) continue;
    for (int dx = 0; dx < 3; ++dx) {
      int xx = 2 * xo + dx - 1; if (xx < 0 || xx >= Wd) continue;
      acc += kF2[dy][dx] * in[((size_t)(n * H + yy) * Wd + xx) * C + c];
    }
  }
  out[idx] = acc * gin[c];
}

// blur stride-1 + pixel_unshuffle(2) + mean over r = 4*Cin/Cout, with pre-gain
__global__ void k_ds_proj(const float* __restrict__ in, const float* __restrict__ gin,
                          float* __restrict__ out, int total, int H, int Wd,
                          int Cin, int Cout)
{
  int idx = blockIdx.x * blockDim.x + threadIdx.x;
  if (idx >= total) return;
  const int Ho = H / 2, Wo = Wd / 2;
  const int r = (4 * Cin) / Cout;
  int co = idx % Cout;
  int t = idx / Cout;
  int xo = t % Wo; t /= Wo;
  int yo = t % Ho;
  int n  = t / Ho;
  float acc = 0.f;
  for (int j = 0; j < r; ++j) {
    int k = co * r + j;
    int c = k >> 2, ry = (k >> 1) & 1, rx = k & 1;
    int yc = 2 * yo + ry, xc = 2 * xo + rx;
    float b = 0.f;
    for (int dy = 0; dy < 3; ++dy) {
      int yy = yc + dy - 1; if (yy < 0 || yy >= H) continue;
      for (int dx = 0; dx < 3; ++dx) {
        int xx = xc + dx - 1; if (xx < 0 || xx >= Wd) continue;
        b += kF2[dy][dx] * in[((size_t)(n * H + yy) * Wd + xx) * Cin + c];
      }
    }
    acc += b * gin[c];
  }
  out[idx] = acc / (float)r;
}

// rope tables: cs/sn [L][32]
__global__ void k_rope(float* __restrict__ cs, float* __restrict__ sn, int H, int Wd)
{
  int idx = blockIdx.x * blockDim.x + threadIdx.x;
  if (idx >= H * Wd * 32) return;
  int i = idx & 31;
  int p = idx >> 5;
  int y = p / Wd, x = p % Wd;
  int j = (i < 16) ? i : i - 16;
  float freq = powf(10000.f, -(float)j / 16.f);
  float ang = ((i < 16) ? (float)y : (float)x) * freq;
  cs[idx] = cosf(ang);
  sn[idx] = sinf(ang);
}

// normalize + rope q,k out of qkv (f32 [M,3C]) into [N,heads,L,64] buffers
__global__ void k_attn_prep(const float* __restrict__ qkv,
                            const float* __restrict__ cs, const float* __restrict__ sn,
                            float* __restrict__ qn, float* __restrict__ kn,
                            int total, int L, int C, int heads)
{
  int idx = blockIdx.x * blockDim.x + threadIdx.x;
  if (idx >= total) return;
  const int h = idx % heads;
  const int tok = idx / heads;
  const int n = tok / L, p = tok % L;
  const float* base = qkv + (size_t)tok * 3 * C;
  const float* cp = cs + (size_t)p * 32;
  const float* sp = sn + (size_t)p * 32;
  for (int which = 0; which < 2; ++which) {
    const float* t = base + which * C + h * 64;
    float ss = 0.f;
    for (int i = 0; i < 64; ++i) ss += t[i] * t[i];
    float rr = rsqrtf(ss + 1e-8f);
    float* ob = (which ? kn : qn) + (((size_t)(n * heads + h)) * L + p) * 64;
    for (int i = 0; i < 32; ++i) {
      float t1 = t[2 * i] * rr, t2 = t[2 * i + 1] * rr;
      ob[2 * i]     = t1 * cp[i] - t2 * sp[i];
      ob[2 * i + 1] = t1 * sp[i] + t2 * cp[i];
    }
  }
}

// attention core: per (n,head) block, one query per thread; sink in denominator
template<int D>
__global__ void k_attn_core(const float* __restrict__ qn, const float* __restrict__ kn,
                            const float* __restrict__ qkv, const float* __restrict__ sinks,
                            half_t* __restrict__ Ob, int L, int C, int heads)
{
  const int nh = blockIdx.x;
  const int n = nh / heads, h = nh % heads;
  const int p = threadIdx.x;
  const float* qp = qn + ((size_t)nh * L + p) * D;
  float q[D];
#pragma unroll
  for (int i = 0; i < D; ++i) q[i] = qp[i];
  const float* kb = kn + (size_t)nh * L * D;
  float m = sinks[h];
  for (int j = 0; j < L; ++j) {
    const float* kj = kb + (size_t)j * D;
    float s = 0.f;
#pragma unroll
    for (int i = 0; i < D; ++i) s += q[i] * kj[i];
    m = fmaxf(m, 8.f * s);
  }
  float Z = expf(sinks[h] - m);
  float o[D];
#pragma unroll
  for (int i = 0; i < D; ++i) o[i] = 0.f;
  const float* vb = qkv + (size_t)n * L * (3 * C) + 2 * C + h * D;
  for (int j = 0; j < L; ++j) {
    const float* kj = kb + (size_t)j * D;
    float s = 0.f;
#pragma unroll
    for (int i = 0; i < D; ++i) s += q[i] * kj[i];
    float e = expf(8.f * s - m);
    Z += e;
    const float* vj = vb + (size_t)j * (3 * C);
#pragma unroll
    for (int i = 0; i < D; ++i) o[i] += e * vj[i];
  }
  float invZ = 1.f / Z;
  half_t* op = Ob + ((size_t)(n * L + p)) * C + h * D;
#pragma unroll
  for (int i = 0; i < D; ++i) op[i] = (half_t)(o[i] * invZ);
}

// final: scale by rsqrt(av), pixel_unshuffle(2) on 2x2, centered 1x1 head conv -> (32,)
__global__ void k_head(const float* __restrict__ X, const float* __restrict__ gin,
                       const float* __restrict__ wc, float* __restrict__ out)
{
  const int n = blockIdx.x;
  __shared__ float red[256];
  float s = 0.f;
  for (int c2 = threadIdx.x; c2 < 2048; c2 += 256) {
    int c = c2 >> 2, ry = (c2 >> 1) & 1, rx = c2 & 1;
    float v = X[(((size_t)n * 2 + ry) * 2 + rx) * 512 + c] * gin[c];
    s += wc[c2] * v;
  }
  red[threadIdx.x] = s; __syncthreads();
  for (int st = 128; st > 0; st >>= 1) {
    if (threadIdx.x < st) red[threadIdx.x] += red[threadIdx.x + st];
    __syncthreads();
  }
  if (threadIdx.x == 0) out[n] = red[0];
}

// ===========================================================================
extern "C" void kernel_launch(void* const* d_in, const int* in_sizes, int n_in,
                              void* d_out, int out_size, void* d_ws, size_t ws_size,
                              hipStream_t stream)
{
  (void)out_size; (void)ws_size;
  if (n_in < 55) return;
  const float* L[55];
  for (int i = 0; i < 55; ++i) L[i] = (const float*)d_in[i];

  const float *Bias, *Gain, *ExtractW, *HeadW, *Xin;
  const float *sAl[4][2];
  const float *sW1[4][2] = {}, *sB1[4][2] = {}, *sW2[4][2] = {}, *sW3[4][2] = {};
  const float *aWqkv[4] = {}, *aBqkv[4] = {}, *aWproj[4] = {}, *aSinks[4] = {};
  const float *hAl[2], *hW1[2], *hB1[2], *hW2[2], *hW3[2];

  if (in_sizes[0] == 3) {
    // jax tree (sorted-dict) flatten order: params first, x last
    Bias = L[0]; ExtractW = L[1]; Gain = L[2]; HeadW = L[3];
    int i = 4;
    for (int r = 0; r < 2; ++r) { // head.res1, head.res2: alpha, b1, w1, w2, w3
      hAl[r] = L[i]; hB1[r] = L[i+1]; hW1[r] = L[i+2]; hW2[r] = L[i+3]; hW3[r] = L[i+4];
      i += 5;
    }
    for (int s = 0; s < 4; ++s) {
      sAl[s][0] = L[i++]; sAl[s][1] = L[i++];
      sB1[s][0] = L[i]; sW1[s][0] = L[i+1]; sW2[s][0] = L[i+2]; sW3[s][0] = L[i+3]; i += 4;
      if (s < 2) { sB1[s][1] = L[i]; sW1[s][1] = L[i+1]; sW2[s][1] = L[i+2]; sW3[s][1] = L[i+3]; i += 4; }
      else { aBqkv[s] = L[i]; aSinks[s] = L[i+1]; aWproj[s] = L[i+2]; aWqkv[s] = L[i+3]; i += 4; }
    }
    Xin = L[54];
  } else {
    // insertion order: x, Bias, Gain, Extract_w, stages..., head(res1,res2,head_w)
    Xin = L[0]; Bias = L[1]; Gain = L[2]; ExtractW = L[3];
    int i = 4;
    for (int s = 0; s < 4; ++s) {
      sAl[s][0] = L[i++]; sAl[s][1] = L[i++];
      sW1[s][0] = L[i]; sB1[s][0] = L[i+1]; sW2[s][0] = L[i+2]; sW3[s][0] = L[i+3]; i += 4;
      if (s < 2) { sW1[s][1] = L[i]; sB1[s][1] = L[i+1]; sW2[s][1] = L[i+2]; sW3[s][1] = L[i+3]; i += 4; }
      else { aWqkv[s] = L[i]; aBqkv[s] = L[i+1]; aWproj[s] = L[i+2]; aSinks[s] = L[i+3]; i += 4; }
    }
    for (int r = 0; r < 2; ++r) {
      hAl[r] = L[i]; hW1[r] = L[i+1]; hB1[r] = L[i+2]; hW2[r] = L[i+3]; hW3[r] = L[i+4];
      i += 5;
    }
    HeadW = L[i];
  }

  // ---- workspace bump allocator ----
  char* base = (char*)d_ws;
  size_t off = 0;
  auto alloc = [&](size_t bytes) -> void* {
    void* p = base + off;
    off += (bytes + 255) & ~(size_t)255;
    return p;
  };
  const int Nb = 32;
  float*  X     = (float*) alloc((size_t)Nb * 64 * 64 * 128 * 4);
  float*  Xb    = (float*) alloc((size_t)Nb * 64 * 64 * 128 * 4);
  half_t* Abuf  = (half_t*)alloc((size_t)Nb * 64 * 64 * 128 * 2);
  half_t* H1    = (half_t*)alloc((size_t)Nb * 64 * 64 * 256 * 2);
  half_t* H2    = (half_t*)alloc((size_t)Nb * 64 * 64 * 256 * 2);
  float*  Ybuf  = (float*) alloc((size_t)Nb * 256 * 1536 * 4);
  float*  qn    = (float*) alloc((size_t)Nb * 8 * 256 * 64 * 4);
  float*  kn    = (float*) alloc((size_t)Nb * 8 * 256 * 64 * 4);
  half_t* Wa    = (half_t*)alloc(2u * 1024 * 1024);
  half_t* Wb    = (half_t*)alloc(2u * 1024 * 1024);
  half_t* Wc    = (half_t*)alloc(2u * 1024 * 1024);
  float*  alpha0 = (float*)alloc(512 * 4);
  float*  alpha1 = (float*)alloc(512 * 4);
  float*  gin1   = (float*)alloc(512 * 4);
  float*  ginds  = (float*)alloc(512 * 4);
  float*  csT    = (float*)alloc(256 * 32 * 4);
  float*  snT    = (float*)alloc(256 * 32 * 4);
  float*  wExt   = (float*)alloc(3456 * 4);
  float*  wHead  = (float*)alloc(2048 * 4);

  auto swapX = [&]() { float* t = X; X = Xb; Xb = t; };

  auto ffn = [&](int M, int H, int Wd, int C, int hid, int cpg,
                 const float* w1, const float* b1, const float* w2, const float* w3,
                 const float* gin, const float* alph)
  {
    int total = M * C;
    k_tohalf<<<(total + 255) / 256, 256, 0, stream>>>(X, gin, Abuf, total, C);
    k_prep_pw<<<hid, 256, 0, stream>>>(w1, nullptr, Wa, C);
    launch_gemm(C, dim3(hid / 64, M / 128), stream,
                Abuf, Wa, b1, nullptr, nullptr, H1, M, hid, 1);
    k_prep_gconv<<<hid, 256, 0, stream>>>(w2, Wb, cpg);
    int tot2 = M * hid;
    k_gconv<<<(tot2 + 255) / 256, 256, 0, stream>>>(H1, Wb, H2, M, H, Wd, hid, cpg);
    k_prep_pw<<<C, 256, 0, stream>>>(w3, alph, Wc, hid);
    launch_gemm(hid, dim3(C / 64, M / 128), stream,
                H2, Wc, nullptr, X, X, nullptr, M, C, 0);
  };

  // ---- extract conv ----
  k_prep_extract<<<128, 32, 0, stream>>>(ExtractW, Gain, wExt);
  {
    int total = Nb * 64 * 64 * 128;
    k_extract<<<(total + 255) / 256, 256, 0, stream>>>(Xin, Bias, wExt, X, total);
  }

  // ---- stages ----
  struct Cfg { int C, H, hid; bool attn; };
  Cfg cfg[4] = { {128, 64, 256, false}, {256, 32, 512, false},
                 {512, 16, 1024, true}, {512, 8, 1024, true} };
  for (int s = 0; s < 4; ++s) {
    const int C = cfg[s].C, H = cfg[s].H, Wd = H, Lx = H * Wd;
    const int M = Nb * Lx, hid = cfg[s].hid;
    k_gins<<<(C + 255) / 256, 256, 0, stream>>>(sAl[s][0], sAl[s][1],
                                                alpha0, alpha1, gin1, ginds, C);
    // block 0: FFN, gin = 1
    ffn(M, H, Wd, C, hid, 16, sW1[s][0], sB1[s][0], sW2[s][0], sW3[s][0], nullptr, alpha0);
    if (!cfg[s].attn) {
      ffn(M, H, Wd, C, hid, 16, sW1[s][1], sB1[s][1], sW2[s][1], sW3[s][1], gin1, alpha1);
    } else {
      const int heads = C / 64;
      int total = M * C;
      k_tohalf<<<(total + 255) / 256, 256, 0, stream>>>(X, gin1, Abuf, total, C);
      k_prep_pw<<<3 * C, 256, 0, stream>>>(aWqkv[s], nullptr, Wa, C);
      launch_gemm(C, dim3(3 * C / 64, M / 128), stream,
                  Abuf, Wa, aBqkv[s], nullptr, Ybuf, nullptr, M, 3 * C, 0);
      k_rope<<<(Lx * 32 + 255) / 256, 256, 0, stream>>>(csT, snT, H, Wd);
      int tp = M * heads;
      k_attn_prep<<<(tp + 255) / 256, 256, 0, stream>>>(Ybuf, csT, snT, qn, kn,
                                                        tp, Lx, C, heads);
      k_attn_core<64><<<Nb * heads, Lx, 0, stream>>>(qn, kn, Ybuf, aSinks[s],
                                                     Abuf, Lx, C, heads);
      k_prep_pw<<<C, 256, 0, stream>>>(aWproj[s], alpha1, Wc, C);
      launch_gemm(C, dim3(C / 64, M / 128), stream,
                  Abuf, Wc, nullptr, X, X, nullptr, M, C, 0);
    }
    // downsample
    if (s == 0) {
      int tot = Nb * 32 * 32 * 256;
      k_ds_proj<<<(tot + 255) / 256, 256, 0, stream>>>(X, ginds, Xb, tot, 64, 64, 128, 256);
      swapX();
    } else if (s == 1) {
      int tot = Nb * 16 * 16 * 512;
      k_ds_proj<<<(tot + 255) / 256, 256, 0, stream>>>(X, ginds, Xb, tot, 32, 32, 256, 512);
      swapX();
    } else if (s == 2) {
      int tot = Nb * 8 * 8 * 512;
      k_ds_same<<<(tot + 255) / 256, 256, 0, stream>>>(X, ginds, Xb, tot, 16, 16, 512);
      swapX();
    } else {
      // head's first downsample uses av from stage 3
      int tot = Nb * 4 * 4 * 512;
      k_ds_same<<<(tot + 255) / 256, 256, 0, stream>>>(X, ginds, Xb, tot, 8, 8, 512);
      swapX();
    }
  }

  // ---- head: res1 @ 4x4 ----
  k_gins<<<2, 256, 0, stream>>>(hAl[0], nullptr, alpha0, alpha1, gin1, ginds, 512);
  ffn(Nb * 16, 4, 4, 512, 1024, 32, hW1[0], hB1[0], hW2[0], hW3[0], nullptr, alpha0);
  {
    int tot = Nb * 2 * 2 * 512;
    k_ds_same<<<(tot + 255) / 256, 256, 0, stream>>>(X, ginds, Xb, tot, 4, 4, 512);
    swapX();
  }
  // ---- head: res2 @ 2x2 ----
  k_gins<<<2, 256, 0, stream>>>(hAl[1], nullptr, alpha0, alpha1, gin1, ginds, 512);
  ffn(Nb * 4, 2, 2, 512, 1024, 32, hW1[1], hB1[1], hW2[1], hW3[1], nullptr, alpha0);

  // ---- head projection ----
  k_prep_headw<<<1, 256, 0, stream>>>(HeadW, wHead);
  k_head<<<Nb, 256, 0, stream>>>(X, ginds, wHead, (float*)d_out);
}
